// BatchGAT_42949672961877
// MI455X (gfx1250) — matrix operand order repaired and checked by reading, not compiled, not run on hardware
//
#include <hip/hip_runtime.h>
#include <hip/hip_bf16.h>
#include <stdint.h>

typedef __attribute__((ext_vector_type(16))) _Float16 v16h;
typedef __attribute__((ext_vector_type(8)))  float    v8f;

#define BS 8
#define NN 1024
#define HEADS 8

union AccU { v8f v; float f[8]; };
union AFragU { v16h v; float4 f4[2]; _Float16 h[16]; };

// ---------------- f32 -> f16 weight conversion ----------------
__global__ void k_f32_to_f16(const float* __restrict__ in, _Float16* __restrict__ out, int n) {
    int i = blockIdx.x * blockDim.x + threadIdx.x;
    if (i < n) out[i] = (_Float16)in[i];
}

// ---------------- embedding gather + concat -> x0 f16 [b,n,128] ----------------
__global__ void k_embed_concat(const float* __restrict__ x, const int* __restrict__ verts,
                               const float* __restrict__ emb_w, _Float16* __restrict__ x0) {
    int row = blockIdx.x;          // b*1024+n  (8192 rows)
    int t   = threadIdx.x;         // 0..127
    float v;
    if (t < 64) v = x[(size_t)row * 64 + t];
    else        v = emb_w[(size_t)verts[row] * 64 + (t - 64)];
    x0[(size_t)row * 128 + t] = (_Float16)v;
}

// ---------------- per-head feature GEMM: HP[b,h,n,F_OUT] = X[b,n,F_IN] @ W[h,F_IN,F_OUT] ----------------
// one wave per 16-row M tile; all F_OUT/16 ntiles computed in-wave sharing the A fragment
template <int F_IN, int F_OUT>
__global__ __launch_bounds__(32) void k_transform(const _Float16* __restrict__ X,
                                                  const _Float16* __restrict__ W,
                                                  _Float16* __restrict__ HP) {
    constexpr int NT = F_OUT / 16;
    int lane  = threadIdx.x;
    int mtile = blockIdx.x;
    int h = blockIdx.y, b = blockIdx.z;
    int hi   = (lane >= 16) ? 8 : 0;           // K-half offset for this lane group
    int rowA = mtile * 16 + (lane & 15);       // A-fragment row (M) for this lane
    const _Float16* xrow = X + (size_t)(b * NN + rowA) * F_IN;

    AccU c[NT];
#pragma unroll
    for (int nt = 0; nt < NT; ++nt) c[nt].v = v8f{};

#pragma unroll
    for (int kb = 0; kb < F_IN; kb += 32) {
        AFragU a;
        a.f4[0] = *reinterpret_cast<const float4*>(xrow + kb + hi);        // K = kb+hi .. +7
        a.f4[1] = *reinterpret_cast<const float4*>(xrow + kb + 16 + hi);   // K = kb+16+hi .. +7
        const _Float16* wrow = W + ((size_t)h * F_IN + kb) * F_OUT + (lane & 15); // B: lane%16 = N col, elements over K
#pragma unroll
        for (int nt = 0; nt < NT; ++nt) {
            v16h bf;
#pragma unroll
            for (int i = 0; i < 16; ++i) bf[i] = wrow[(size_t)((i < 8) ? hi + i : 16 + hi + i - 8) * F_OUT + nt * 16];
            c[nt].v = __builtin_amdgcn_wmma_f32_16x16x32_f16(false, a.v, false, bf, (short)0,
                                                             c[nt].v, false, false);
        }
    }
    // D layout: lane%16 = N col, VGPR g = M row g(+8 for high lanes)
    _Float16* out = HP + (size_t)(b * HEADS + h) * NN * F_OUT;
#pragma unroll
    for (int nt = 0; nt < NT; ++nt) {
        int col = (lane & 15) + nt * 16;
#pragma unroll
        for (int g = 0; g < 8; ++g) {
            int row = mtile * 16 + g + hi;
            out[(size_t)row * F_OUT + col] = (_Float16)c[nt].f[g];
        }
    }
}

// ---------------- s,d projections: s = tanh(h') . a_src, d = tanh(h') . a_dst ----------------
template <int F_OUT>
__global__ void k_sd(const _Float16* __restrict__ HP, const float* __restrict__ a_src,
                     const float* __restrict__ a_dst, float* __restrict__ S, float* __restrict__ D) {
    int idx = blockIdx.x * blockDim.x + threadIdx.x;   // (b*8+h)*1024+n
    if (idx >= BS * HEADS * NN) return;
    int h = (idx >> 10) & 7;
    const _Float16* hp = HP + (size_t)idx * F_OUT;
    float s = 0.f, d = 0.f;
#pragma unroll
    for (int o = 0; o < F_OUT; ++o) {
        float t = tanhf((float)hp[o]);
        s += t * a_src[h * F_OUT + o];
        d += t * a_dst[h * F_OUT + o];
    }
    S[idx] = s; D[idx] = d;
}

// ---------------- masked softmax stats: two-sweep (max, then single-exp sum) ----------------
// one wave per row; each lane handles 4 consecutive columns per step (float4/uchar4 loads)
__global__ void k_softmax_stats(const float* __restrict__ S, const float* __restrict__ D,
                                const unsigned char* __restrict__ adj,
                                float* __restrict__ M, float* __restrict__ Rinv) {
    int lane = threadIdx.x & 31;
    int wid  = threadIdx.x >> 5;
    int row  = blockIdx.x * 8 + wid;       // (b*8+h)*1024+n, 65536 rows
    int b = row >> 13;
    int n = row & 1023;
    float s = S[row];
    const float* d = D + (row & ~1023);
    const unsigned char* arow = adj + ((size_t)b * NN + n) * NN;

    // sweep 1: masked max (no transcendentals)
    float m = -1e30f;
    for (int base = 0; base < NN; base += 128) {
        int j = base + lane * 4;
        float4 dv = *reinterpret_cast<const float4*>(d + j);
        union { unsigned int u; unsigned char cb[4]; } mk;
        mk.u = *reinterpret_cast<const unsigned int*>(arow + j);
        float dvv[4] = {dv.x, dv.y, dv.z, dv.w};
#pragma unroll
        for (int t = 0; t < 4; ++t) {
            float sc = s + dvv[t];
            sc = sc >= 0.f ? sc : 0.2f * sc;
            m = mk.cb[t] ? fmaxf(m, sc) : m;
        }
    }
#pragma unroll
    for (int off = 16; off; off >>= 1) m = fmaxf(m, __shfl_xor(m, off, 32));

    // sweep 2: one exp per valid element
    float l = 0.f;
    for (int base = 0; base < NN; base += 128) {
        int j = base + lane * 4;
        float4 dv = *reinterpret_cast<const float4*>(d + j);
        union { unsigned int u; unsigned char cb[4]; } mk;
        mk.u = *reinterpret_cast<const unsigned int*>(arow + j);
        float dvv[4] = {dv.x, dv.y, dv.z, dv.w};
#pragma unroll
        for (int t = 0; t < 4; ++t) {
            float sc = s + dvv[t];
            sc = sc >= 0.f ? sc : 0.2f * sc;
            l += mk.cb[t] ? __expf(sc - m) : 0.f;
        }
    }
#pragma unroll
    for (int off = 16; off; off >>= 1) l += __shfl_xor(l, off, 32);

    if (lane == 0) { M[row] = m; Rinv[row] = 1.f / l; }
}

// ---------------- attention apply: OUT = softmax(P) @ h' (+bias, optional ELU->x1 f16) ----------------
// one wave per 16-row M tile; probabilities generated ONCE per chunk into the WMMA A-fragment
// layout and reused for all F_OUT/16 ntiles (separate accumulators).
template <int F_OUT, bool ELU_TO_X1>
__global__ __launch_bounds__(32) void k_attn_apply(const float* __restrict__ S, const float* __restrict__ D,
                                                   const float* __restrict__ M, const float* __restrict__ Rinv,
                                                   const unsigned char* __restrict__ adj,
                                                   const _Float16* __restrict__ HP,
                                                   const float* __restrict__ bias,
                                                   _Float16* __restrict__ X1, float* __restrict__ AO) {
    constexpr int NT = F_OUT / 16;
    int lane  = threadIdx.x;
    int mtile = blockIdx.x;
    int h = blockIdx.y, b = blockIdx.z;
    int bh = b * HEADS + h;
    int hi   = (lane >= 16) ? 8 : 0;
    int rowA = mtile * 16 + (lane & 15);
    int gr   = bh * NN + rowA;
    float s = S[gr], rm = M[gr], rinv = Rinv[gr];
    const float* dptr = D + (size_t)bh * NN;
    const unsigned char* arow = adj + ((size_t)b * NN + rowA) * NN;
    const _Float16* hpb = HP + (size_t)bh * NN * F_OUT;

    AccU c[NT];
#pragma unroll
    for (int nt = 0; nt < NT; ++nt) c[nt].v = v8f{};

    for (int kb = 0; kb < NN; kb += 32) {
        // d values for this lane's 16 K-columns: two contiguous runs of 8
        union { float4 f4[4]; float x[16]; } dd;
        dd.f4[0] = *reinterpret_cast<const float4*>(dptr + kb + hi);
        dd.f4[1] = *reinterpret_cast<const float4*>(dptr + kb + hi + 4);
        dd.f4[2] = *reinterpret_cast<const float4*>(dptr + kb + 16 + hi);
        dd.f4[3] = *reinterpret_cast<const float4*>(dptr + kb + 16 + hi + 4);
        union { uint2 u; unsigned char cb[8]; } mk0, mk1;
        mk0.u = *reinterpret_cast<const uint2*>(arow + kb + hi);
        mk1.u = *reinterpret_cast<const uint2*>(arow + kb + 16 + hi);

        AFragU a;
#pragma unroll
        for (int i = 0; i < 16; ++i) {
            unsigned char mb = (i < 8) ? mk0.cb[i] : mk1.cb[i - 8];
            float sc = s + dd.x[i];
            sc = sc >= 0.f ? sc : 0.2f * sc;
            float p = mb ? __expf(sc - rm) * rinv : 0.f;
            a.h[i] = (_Float16)p;
        }
        const _Float16* brow = hpb + (size_t)kb * F_OUT + (lane & 15);  // B: lane%16 = N col, elements over K (attended col)
#pragma unroll
        for (int nt = 0; nt < NT; ++nt) {
            v16h bf;
#pragma unroll
            for (int i = 0; i < 16; ++i) bf[i] = brow[(size_t)((i < 8) ? hi + i : 16 + hi + i - 8) * F_OUT + nt * 16];
            c[nt].v = __builtin_amdgcn_wmma_f32_16x16x32_f16(false, a.v, false, bf, (short)0,
                                                             c[nt].v, false, false);
        }
    }

#pragma unroll
    for (int nt = 0; nt < NT; ++nt) {
        int colf = (lane & 15) + nt * 16;
        float bv = bias[colf];
#pragma unroll
        for (int g = 0; g < 8; ++g) {
            int row = mtile * 16 + g + hi;
            float v = c[nt].f[g] + bv;
            if (ELU_TO_X1) {
                v = v > 0.f ? v : (__expf(v) - 1.f);                             // ELU(alpha=1)
                X1[((size_t)b * NN + row) * 256 + h * 32 + colf] = (_Float16)v;  // head-interleaved
            } else {
                AO[((size_t)bh * NN + row) * 16 + colf] = v;
            }
        }
    }
}

// ---------------- head mean + log_softmax over 16 features ----------------
__global__ void k_final(const float* __restrict__ AO, float* __restrict__ OUT) {
    int idx = blockIdx.x * blockDim.x + threadIdx.x;   // b*1024+n, 8192
    if (idx >= BS * NN) return;
    int b = idx >> 10, n = idx & 1023;
    float v[16];
#pragma unroll
    for (int o = 0; o < 16; ++o) v[o] = 0.f;
    for (int h = 0; h < HEADS; ++h) {
        const float* p = AO + ((size_t)(b * HEADS + h) * NN + n) * 16;
#pragma unroll
        for (int o = 0; o < 16; ++o) v[o] += p[o];
    }
    float mx = -1e30f;
#pragma unroll
    for (int o = 0; o < 16; ++o) { v[o] *= 0.125f; mx = fmaxf(mx, v[o]); }
    float se = 0.f;
#pragma unroll
    for (int o = 0; o < 16; ++o) se += __expf(v[o] - mx);
    float lse = __logf(se);
#pragma unroll
    for (int o = 0; o < 16; ++o) OUT[(size_t)idx * 16 + o] = v[o] - mx - lse;
}

extern "C" void kernel_launch(void* const* d_in, const int* in_sizes, int n_in,
                              void* d_out, int out_size, void* d_ws, size_t ws_size,
                              hipStream_t stream) {
    const float*         x      = (const float*)d_in[0];
    const int*           verts  = (const int*)d_in[1];
    const unsigned char* adj    = (const unsigned char*)d_in[2];
    const float*         emb_w  = (const float*)d_in[3];
    const float*         w0     = (const float*)d_in[4];
    const float*         a_src0 = (const float*)d_in[5];
    const float*         a_dst0 = (const float*)d_in[6];
    const float*         b0     = (const float*)d_in[7];
    const float*         w1     = (const float*)d_in[8];
    const float*         a_src1 = (const float*)d_in[9];
    const float*         a_dst1 = (const float*)d_in[10];
    const float*         b1     = (const float*)d_in[11];
    float* out = (float*)d_out;

    // workspace carve-up (all 256B aligned)
    char* ws = (char*)d_ws;
    size_t off = 0;
    auto alloc = [&](size_t bytes) { char* p = ws + off; off = (off + bytes + 255) & ~(size_t)255; return p; };
    _Float16* x0f16  = (_Float16*)alloc((size_t)BS * NN * 128 * 2);
    _Float16* w0f16  = (_Float16*)alloc((size_t)HEADS * 128 * 32 * 2);
    _Float16* w1f16  = (_Float16*)alloc((size_t)HEADS * 256 * 16 * 2);
    _Float16* hp0    = (_Float16*)alloc((size_t)BS * HEADS * NN * 32 * 2);
    float*    s0     = (float*)alloc((size_t)BS * HEADS * NN * 4);
    float*    d0     = (float*)alloc((size_t)BS * HEADS * NN * 4);
    float*    m0     = (float*)alloc((size_t)BS * HEADS * NN * 4);
    float*    r0     = (float*)alloc((size_t)BS * HEADS * NN * 4);
    _Float16* x1f16  = (_Float16*)alloc((size_t)BS * NN * 256 * 2);
    _Float16* hp1    = (_Float16*)alloc((size_t)BS * HEADS * NN * 16 * 2);
    float*    s1     = (float*)alloc((size_t)BS * HEADS * NN * 4);
    float*    d1     = (float*)alloc((size_t)BS * HEADS * NN * 4);
    float*    m1     = (float*)alloc((size_t)BS * HEADS * NN * 4);
    float*    r1     = (float*)alloc((size_t)BS * HEADS * NN * 4);
    float*    ao1    = (float*)alloc((size_t)BS * HEADS * NN * 16 * 4);
    (void)ws_size; (void)n_in; (void)in_sizes; (void)out_size;

    // weight conversion
    k_f32_to_f16<<<(HEADS * 128 * 32 + 255) / 256, 256, 0, stream>>>(w0, w0f16, HEADS * 128 * 32);
    k_f32_to_f16<<<(HEADS * 256 * 16 + 255) / 256, 256, 0, stream>>>(w1, w1f16, HEADS * 256 * 16);

    // embedding + concat
    k_embed_concat<<<BS * NN, 128, 0, stream>>>(x, verts, emb_w, x0f16);

    // ---- layer 0 ----
    k_transform<128, 32><<<dim3(64, HEADS, BS), 32, 0, stream>>>(x0f16, w0f16, hp0);
    k_sd<32><<<(BS * HEADS * NN + 255) / 256, 256, 0, stream>>>(hp0, a_src0, a_dst0, s0, d0);
    k_softmax_stats<<<BS * HEADS * NN / 8, 256, 0, stream>>>(s0, d0, adj, m0, r0);
    k_attn_apply<32, true><<<dim3(64, HEADS, BS), 32, 0, stream>>>(s0, d0, m0, r0, adj, hp0, b0,
                                                                   x1f16, nullptr);
    // ---- layer 1 ----
    k_transform<256, 16><<<dim3(64, HEADS, BS), 32, 0, stream>>>(x1f16, w1f16, hp1);
    k_sd<16><<<(BS * HEADS * NN + 255) / 256, 256, 0, stream>>>(hp1, a_src1, a_dst1, s1, d1);
    k_softmax_stats<<<BS * HEADS * NN / 8, 256, 0, stream>>>(s1, d1, adj, m1, r1);
    k_attn_apply<16, false><<<dim3(64, HEADS, BS), 32, 0, stream>>>(s1, d1, m1, r1, adj, hp1, b1,
                                                                    nullptr, ao1);
    // ---- head mean + log_softmax ----
    k_final<<<(BS * NN + 255) / 256, 256, 0, stream>>>(ao1, out);
}
